// CapsuleLayer_5952824672874
// MI455X (gfx1250) — matrix operand order repaired and checked by reading, not compiled, not run on hardware
//
#include <hip/hip_runtime.h>
#include <hip/hip_bf16.h>

// Problem constants (from reference): x[B,N,I] f32, W[M,N,D,I] f32 -> out[B,M,D] f32
#define B_ 128
#define N_ 512
#define I_ 256
#define M_ 32
#define D_ 16

typedef __attribute__((ext_vector_type(16))) __bf16 v16bf;
typedef __attribute__((ext_vector_type(8)))  float  v8f;
typedef __attribute__((ext_vector_type(4)))  float  v4f;

// ---------------------------------------------------------------------------
// Kernel 1: inputs_hat[b,m,n,d] = sum_i x[b,n,i] * W[m,n,d,i]
// One wave per 16(batch)x16(D) tile, K=I contracted in steps of 32 via
// v_wmma_f32_16x16x32_bf16. Workgroup = (n, 8 consecutive m); 8 waves = 8
// batch tiles. W slice staged to LDS as bf16 (double-buffered: stage k+1
// while computing k, single barrier per k-step). B fragments are prefetched
// one mq ahead so ds_load latency overlaps the WMMA pipe.
// hat stored bf16 in [M,N,B,D] layout.
// ---------------------------------------------------------------------------
__global__ __launch_bounds__(256) void caps_gemm_hat(
    const float* __restrict__ x, const float* __restrict__ W,
    __bf16* __restrict__ hat)
{
    __shared__ __bf16 wlds[2][8 * 16 * 32];   // 2 x [m'][d][k] bf16, 2 x 8 KB

    const int n     = blockIdx.x;          // 0..511
    const int mbase = blockIdx.y * 8;      // 0,8,16,24
    const int t     = threadIdx.x;
    const int lane  = t & 31;
    const int wave  = t >> 5;              // 0..7 -> batch tile
    const int bbase = wave * 16;
    const int col   = lane & 15;           // A row / B col / C col
    const int half  = lane >> 4;           // lane half selects K sub-range

    v8f acc[8] = {};                       // 8 m-accumulators, f32

    // A source: x rows for this wave's batch tile (contiguous in i)
    const float* xrow = x + (size_t)(bbase + col) * (N_ * I_) + (size_t)n * I_;

    // W -> LDS staging map: 2 threads per (m',d), each stages 16 k (64B f32)
    const int p  = t >> 1;
    const int sm = p >> 4;                 // m' 0..7
    const int sd = p & 15;                 // d  0..15
    const int sk = (t & 1) * 16;           // k sub-chunk 0 or 16
    const float* wsrc = W + (size_t)(mbase + sm) * (N_ * D_ * I_)
                          + (size_t)n * (D_ * I_) + (size_t)sd * I_ + sk;
    const int wdstoff = (sm * 16 + sd) * 32 + sk;

    // ---- prologue: stage k0 = 0 into buffer 0 ----
    {
        v4f f0 = *(const v4f*)(wsrc + 0);
        v4f f1 = *(const v4f*)(wsrc + 4);
        v4f f2 = *(const v4f*)(wsrc + 8);
        v4f f3 = *(const v4f*)(wsrc + 12);
        v16bf wv;
#pragma unroll
        for (int j = 0; j < 4; ++j) {
            wv[j]      = (__bf16)f0[j];
            wv[4 + j]  = (__bf16)f1[j];
            wv[8 + j]  = (__bf16)f2[j];
            wv[12 + j] = (__bf16)f3[j];
        }
        *(v16bf*)&wlds[0][wdstoff] = wv;
    }

    for (int ks = 0; ks < I_ / 32; ++ks) {
        const int k0 = ks * 32;
        __syncthreads();   // buffer ks&1 fully staged; buffer (ks+1)&1 free

        // ---- stage next W tile into the other buffer (overlaps compute) ----
        if (ks + 1 < I_ / 32) {
            v4f f0 = *(const v4f*)(wsrc + k0 + 32 + 0);
            v4f f1 = *(const v4f*)(wsrc + k0 + 32 + 4);
            v4f f2 = *(const v4f*)(wsrc + k0 + 32 + 8);
            v4f f3 = *(const v4f*)(wsrc + k0 + 32 + 12);
            v16bf wv;
#pragma unroll
            for (int j = 0; j < 4; ++j) {
                wv[j]      = (__bf16)f0[j];
                wv[4 + j]  = (__bf16)f1[j];
                wv[8 + j]  = (__bf16)f2[j];
                wv[12 + j] = (__bf16)f3[j];
            }
            *(v16bf*)&wlds[(ks + 1) & 1][wdstoff] = wv;
        }

        // ---- load A fragment (16x32 of x), A-layout used here (same k order as B below):
        // lane holds row=col; elems 0..15 -> K=klo..klo+15 (contiguous),
        // klo = 0 (lanes 0-15) or 16 (lanes 16-31).
        v16bf a;
        {
            const int klo = half * 16;
            v4f a0 = *(const v4f*)(xrow + k0 + klo + 0);
            v4f a1 = *(const v4f*)(xrow + k0 + klo + 4);
            v4f a2 = *(const v4f*)(xrow + k0 + klo + 8);
            v4f a3 = *(const v4f*)(xrow + k0 + klo + 12);
#pragma unroll
            for (int j = 0; j < 4; ++j) {
                a[j]      = (__bf16)a0[j];
                a[4 + j]  = (__bf16)a1[j];
                a[8 + j]  = (__bf16)a2[j];
                a[12 + j] = (__bf16)a3[j];
            }
        }

        // ---- 8 WMMAs; B fragment for mq+1 prefetched while mq computes.
        // B-layout: lane = col d, K = half*16 + e (e = 0..15), contiguous. ----
        const __bf16* lb = &wlds[ks & 1][col * 32 + half * 16];
        v16bf bcur = *(const v16bf*)lb;
#pragma unroll
        for (int mq = 0; mq < 8; ++mq) {
            v16bf bnext;
            if (mq < 7) bnext = *(const v16bf*)(lb + (mq + 1) * (16 * 32));
            acc[mq] = __builtin_amdgcn_wmma_f32_16x16x32_bf16(
                false, a, false, bcur, (short)0, acc[mq], false, false);
            bcur = bnext;
        }
    }

    // ---- store C tiles: lane holds col d = col, rows v + 8*half ----
#pragma unroll
    for (int mq = 0; mq < 8; ++mq) {
        const size_t base = (((size_t)(mbase + mq) * N_ + n) * B_) * D_;
#pragma unroll
        for (int v = 0; v < 8; ++v) {
            const int r = bbase + v + 8 * half;
            hat[base + (size_t)r * D_ + col] = (__bf16)acc[mq][v];
        }
    }
}

// ---------------------------------------------------------------------------
// Kernel 2: c[b,n,:] = softmax over m of bl[b,n,:]  (layout [B,N,M], m fastest)
// uniform==1: iteration 0, c = 1/32 without reading bl.
// ---------------------------------------------------------------------------
__global__ void caps_softmax(const float* __restrict__ bl,
                             float* __restrict__ c, int uniform)
{
    const int g = blockIdx.x * blockDim.x + threadIdx.x;   // (b,n) pair
    if (g >= B_ * N_) return;
    float* cp = c + (size_t)g * M_;
    if (uniform) {
        const v4f u = {0.03125f, 0.03125f, 0.03125f, 0.03125f};
#pragma unroll
        for (int i = 0; i < 8; ++i) *(v4f*)(cp + i * 4) = u;
        return;
    }
    const float* bp = bl + (size_t)g * M_;
    float vals[M_];
    float mx = -3.0e38f;
#pragma unroll
    for (int i = 0; i < 8; ++i) {
        v4f v = *(const v4f*)(bp + i * 4);
#pragma unroll
        for (int j = 0; j < 4; ++j) {
            vals[i * 4 + j] = v[j];
            mx = fmaxf(mx, v[j]);
        }
    }
    float s = 0.f;
#pragma unroll
    for (int i = 0; i < M_; ++i) { float e = __expf(vals[i] - mx); vals[i] = e; s += e; }
    const float inv = 1.0f / s;
#pragma unroll
    for (int i = 0; i < 8; ++i) {
        v4f v;
#pragma unroll
        for (int j = 0; j < 4; ++j) v[j] = vals[i * 4 + j] * inv;
        *(v4f*)(cp + i * 4) = v;
    }
}

// ---------------------------------------------------------------------------
// Kernel 3: s[b,m,d] = sum_n c[b,n,m] * hat[m,n,b,d]; out = squash(s).
// One block per (b,m); thread t: d = t&15, n-group = t>>4.
// ---------------------------------------------------------------------------
__global__ __launch_bounds__(256) void caps_sout(
    const float* __restrict__ c, const __bf16* __restrict__ hat,
    float* __restrict__ out)
{
    __shared__ float red[256];
    __shared__ float sv[D_];
    const int bm = blockIdx.x;             // 0..B*M-1
    const int b  = bm >> 5;
    const int m  = bm & 31;
    const int t  = threadIdx.x;
    const int d  = t & 15;
    const int ng = t >> 4;

    const __bf16* hp = hat + ((size_t)m * N_) * (B_ * D_) + (size_t)b * D_ + d;
    const float*  cp = c + ((size_t)b * N_) * M_ + m;

    float acc = 0.f;
    for (int n = ng; n < N_; n += 16) {
        const float hv = (float)hp[(size_t)n * (B_ * D_)];
        const float cv = cp[(size_t)n * M_];
        acc = fmaf(cv, hv, acc);
    }
    red[t] = acc;
    __syncthreads();
    if (t < D_) {
        float s = 0.f;
#pragma unroll
        for (int g2 = 0; g2 < 16; ++g2) s += red[g2 * 16 + t];
        sv[t] = s;
    }
    __syncthreads();
    if (t < D_) {
        float s2 = 0.f;
#pragma unroll
        for (int j = 0; j < D_; ++j) { const float q = sv[j]; s2 = fmaf(q, q, s2); }
        const float scale = s2 / (1.0f + s2) * rsqrtf(s2 + 1e-7f);
        out[(size_t)bm * D_ + t] = scale * sv[t];
    }
}

// ---------------------------------------------------------------------------
// Kernel 4: bl[b,n,m] (+)= sum_d out[b,m,d] * hat[m,n,b,d]
// One thread per (b,n,m), m fastest.
// ---------------------------------------------------------------------------
__global__ void caps_bupdate(const float* __restrict__ out,
                             const __bf16* __restrict__ hat,
                             float* __restrict__ bl, int store)
{
    const int g = blockIdx.x * blockDim.x + threadIdx.x;
    if (g >= B_ * N_ * M_) return;
    const int m = g & 31;
    const int n = (g >> 5) & 511;
    const int b = g >> 14;
    const __bf16* hp = hat + (((size_t)m * N_ + n) * B_ + b) * D_;
    const float*  op = out + ((size_t)b * M_ + m) * D_;
    float dot = 0.f;
#pragma unroll
    for (int d2 = 0; d2 < D_; ++d2) dot = fmaf((float)hp[d2], op[d2], dot);
    if (store) bl[g] = dot;
    else       bl[g] += dot;
}

// ---------------------------------------------------------------------------
extern "C" void kernel_launch(void* const* d_in, const int* in_sizes, int n_in,
                              void* d_out, int out_size, void* d_ws, size_t ws_size,
                              hipStream_t stream)
{
    const float* x = (const float*)d_in[0];   // [B,N,I]
    const float* W = (const float*)d_in[1];   // [M,N,D,I]
    float* out = (float*)d_out;               // [B,M,D]

    char* ws = (char*)d_ws;
    __bf16* hat = (__bf16*)ws;                                      // 64 MB
    float*  c   = (float*)(ws + (size_t)B_ * M_ * N_ * D_ * 2);     // 8 MB
    float*  bl  = (float*)((char*)c + (size_t)B_ * N_ * M_ * 4);    // 8 MB
    float*  ot  = (float*)((char*)bl + (size_t)B_ * N_ * M_ * 4);   // 256 KB

    // inputs_hat via WMMA (bf16 A/B, f32 accumulate)
    caps_gemm_hat<<<dim3(N_, M_ / 8), 256, 0, stream>>>(x, W, hat);

    // routing iteration 0 (uniform coupling)
    caps_softmax<<<(B_ * N_) / 256, 256, 0, stream>>>(bl, c, 1);
    caps_sout<<<B_ * M_, 256, 0, stream>>>(c, hat, ot);
    caps_bupdate<<<(B_ * N_ * M_) / 256, 256, 0, stream>>>(ot, hat, bl, 1);

    // iteration 1
    caps_softmax<<<(B_ * N_) / 256, 256, 0, stream>>>(bl, c, 0);
    caps_sout<<<B_ * M_, 256, 0, stream>>>(c, hat, ot);
    caps_bupdate<<<(B_ * N_ * M_) / 256, 256, 0, stream>>>(ot, hat, bl, 0);

    // iteration 2 (final squash -> d_out)
    caps_softmax<<<(B_ * N_) / 256, 256, 0, stream>>>(bl, c, 0);
    caps_sout<<<B_ * M_, 256, 0, stream>>>(c, hat, out);
}
